// SwinTransformerBlock_10763188043922
// MI455X (gfx1250) — matrix operand order repaired and checked by reading, not compiled, not run on hardware
//
#include <hip/hip_runtime.h>
#include <hip/hip_bf16.h>
#include <math.h>

// ---------------- problem constants ----------------
#define BB   16
#define HH   56
#define CC   512
#define WSZ  7
#define SSH  3
#define NHD  16
#define HD   32
#define MLPD 2048
#define NTOK 49            // real tokens per window
#define NPAD 64            // padded tokens per window
#define NWIN 64            // windows per image (8x8)
#define NW_TOT (BB*NWIN)   // 1024 windows
#define MPAD (NW_TOT*NPAD) // 65536 padded rows
#define MDEN (BB*HH*HH)    // 50176 dense rows
#define SCALE 0.17677669529663689f  // 32^-0.5

typedef _Float16 half8  __attribute__((ext_vector_type(8)));
typedef _Float16 v16h   __attribute__((ext_vector_type(16)));
typedef float    v8f    __attribute__((ext_vector_type(8)));

static __device__ __forceinline__ v16h cat16(half8 a, half8 b) {
  return __builtin_shufflevector(a, b, 0,1,2,3,4,5,6,7,8,9,10,11,12,13,14,15);
}
// B fragment / contiguous 32B: halves = p[0..15]
static __device__ __forceinline__ v16h ld_b32B(const _Float16* p) {
  half8 a = *(const half8*)p; half8 b = *(const half8*)(p + 8);
  return cat16(a, b);
}
// A fragment: halves 0..7 = p[0..7], halves 8..15 = p[8..15]
static __device__ __forceinline__ v16h ld_afrag(const _Float16* p) {
  half8 a = *(const half8*)p; half8 b = *(const half8*)(p + 8);
  return cat16(a, b);
}
static __device__ __forceinline__ v8f wmma_f16(v16h a, v16h b, v8f c) {
  return __builtin_amdgcn_wmma_f32_16x16x32_f16(false, a, false, b, (short)0, c, false, false);
}

// ---------------- weight cast + transpose (K x N f32 -> N x K f16) ----------------
__global__ void transpose_cast(const float* __restrict__ src, _Float16* __restrict__ dst,
                               int K, int N) {
  int idx = blockIdx.x * 256 + threadIdx.x;
  if (idx >= K * N) return;
  int n = idx / K, k = idx - n * K;
  dst[idx] = (_Float16)src[(size_t)k * N + n];
}

// ---------------- LN1 + cyclic shift + window partition + pad, f32 -> f16 ----------------
__global__ __launch_bounds__(256) void ln_shift_win(
    const float* __restrict__ x, const float* __restrict__ g, const float* __restrict__ be,
    _Float16* __restrict__ xw) {
  int row = blockIdx.x * 8 + (threadIdx.x >> 5);
  int L = threadIdx.x & 31;
  _Float16* dst = xw + (size_t)row * CC + L * 16;
  int w = row >> 6, m = row & 63;
  if (m >= NTOK) { half8 z = {}; *(half8*)dst = z; *(half8*)(dst + 8) = z; return; }
  int wl = w & 63, b = w >> 6;
  int mr = m / WSZ, mc = m - mr * WSZ;
  int gr = (wl >> 3) * WSZ + mr, gc = (wl & 7) * WSZ + mc;   // shifted coords
  int hh = gr + SSH; if (hh >= HH) hh -= HH;                 // roll(-SS): src = (g+SS)%H
  int ww = gc + SSH; if (ww >= HH) ww -= HH;
  const float* src = x + ((size_t)b * (HH*HH) + hh * HH + ww) * CC + L * 16;
  float v[16]; float s = 0.f, q = 0.f;
  #pragma unroll
  for (int j = 0; j < 4; j++) {
    float4 t = *(const float4*)(src + j * 4);
    v[j*4+0]=t.x; v[j*4+1]=t.y; v[j*4+2]=t.z; v[j*4+3]=t.w;
  }
  #pragma unroll
  for (int j = 0; j < 16; j++) { s += v[j]; q += v[j] * v[j]; }
  #pragma unroll
  for (int o = 16; o > 0; o >>= 1) { s += __shfl_xor(s, o, 32); q += __shfl_xor(q, o, 32); }
  float mean = s * (1.0f / CC);
  float rstd = rsqrtf(q * (1.0f / CC) - mean * mean + 1e-6f);
  #pragma unroll
  for (int j = 0; j < 16; j++) {
    int c = L * 16 + j;
    dst[j] = (_Float16)((v[j] - mean) * rstd * g[c] + be[c]);
  }
}

// ---------------- LN2 (dense rows), f32 -> f16 ----------------
__global__ __launch_bounds__(256) void ln_dense(
    const float* __restrict__ x2, const float* __restrict__ g, const float* __restrict__ be,
    _Float16* __restrict__ h2) {
  int row = blockIdx.x * 8 + (threadIdx.x >> 5);
  int L = threadIdx.x & 31;
  const float* src = x2 + (size_t)row * CC + L * 16;
  _Float16* dst = h2 + (size_t)row * CC + L * 16;
  float v[16]; float s = 0.f, q = 0.f;
  #pragma unroll
  for (int j = 0; j < 4; j++) {
    float4 t = *(const float4*)(src + j * 4);
    v[j*4+0]=t.x; v[j*4+1]=t.y; v[j*4+2]=t.z; v[j*4+3]=t.w;
  }
  #pragma unroll
  for (int j = 0; j < 16; j++) { s += v[j]; q += v[j] * v[j]; }
  #pragma unroll
  for (int o = 16; o > 0; o >>= 1) { s += __shfl_xor(s, o, 32); q += __shfl_xor(q, o, 32); }
  float mean = s * (1.0f / CC);
  float rstd = rsqrtf(q * (1.0f / CC) - mean * mean + 1e-6f);
  #pragma unroll
  for (int j = 0; j < 16; j++) {
    int c = L * 16 + j;
    dst[j] = (_Float16)((v[j] - mean) * rstd * g[c] + be[c]);
  }
}

// ---------------- WMMA GEMM core: 32x64 strip per wave, copy-free ping-pong ----------------
// X: M x K row-major f16; Wt: N x K row-major f16 (pre-transposed weights)
// acc[mt][nt]: rows m0 + mt*16 + r + hi*8, cols n0 + nt*16 + lane%16
// K must be a multiple of 64 (512 or 2048 here).
struct Frag { v16h a0, a1, b[4]; };

static __device__ __forceinline__ void frag_load(
    Frag& f, const _Float16* xr0, const _Float16* xr1,
    const _Float16* wr, size_t wstride, int k0) {
  f.a0 = ld_afrag(xr0 + k0);
  f.a1 = ld_afrag(xr1 + k0);
  #pragma unroll
  for (int nt = 0; nt < 4; nt++) f.b[nt] = ld_b32B(wr + nt * wstride + k0);
}
static __device__ __forceinline__ void frag_mma(Frag& f, v8f acc[2][4]) {
  #pragma unroll
  for (int nt = 0; nt < 4; nt++) {
    acc[0][nt] = wmma_f16(f.a0, f.b[nt], acc[0][nt]);
    acc[1][nt] = wmma_f16(f.a1, f.b[nt], acc[1][nt]);
  }
}

static __device__ __forceinline__ void gemm_core(
    const _Float16* __restrict__ X, const _Float16* __restrict__ Wt,
    int K, int m0, int n0, int lc, int hi, v8f acc[2][4]) {
  #pragma unroll
  for (int mt = 0; mt < 2; mt++)
    #pragma unroll
    for (int nt = 0; nt < 4; nt++)
      #pragma unroll
      for (int r = 0; r < 8; r++) acc[mt][nt][r] = 0.0f;
  const _Float16* xr0 = X + (size_t)(m0 + lc) * K + hi * 16;
  const _Float16* xr1 = X + (size_t)(m0 + 16 + lc) * K + hi * 16;
  const _Float16* wr  = Wt + (size_t)(n0 + lc) * K + hi * 16;
  const size_t wstride = (size_t)16 * K;           // between nt tiles
  Frag fA, fB;                                     // persistent ping-pong homes
  frag_load(fA, xr0, xr1, wr, wstride, 0);
  frag_load(fB, xr0, xr1, wr, wstride, 32);
  for (int k = 64; k + 64 <= K; k += 64) {
    frag_mma(fA, acc);                             // consume A @ (k-64)
    frag_load(fA, xr0, xr1, wr, wstride, k);       // refill A, covered by B's burst
    frag_mma(fB, acc);                             // consume B @ (k-32)
    frag_load(fB, xr0, xr1, wr, wstride, k + 32);  // refill B, covered by A's burst
  }
  frag_mma(fA, acc);                               // A @ (K-64)
  frag_mma(fB, acc);                               // B @ (K-32)
}

// ---------------- GEMM: QKV (M=65536, K=512, N=1536) ----------------
__global__ __launch_bounds__(128) void gemm_qkv(
    const _Float16* __restrict__ X, const _Float16* __restrict__ Wt,
    const float* __restrict__ bias,
    _Float16* __restrict__ qh, _Float16* __restrict__ kh, _Float16* __restrict__ vth) {
  int wv = threadIdx.x >> 5, L = threadIdx.x & 31, lc = L & 15, hi = L >> 4;
  int m0 = blockIdx.x * 128 + wv * 32, n0 = blockIdx.y * 64;
  v8f acc[2][4];
  gemm_core(X, Wt, CC, m0, n0, lc, hi, acc);
  #pragma unroll
  for (int nt = 0; nt < 4; nt++) {
    int c = n0 + nt * 16 + lc;
    float bia = bias[c];
    int which = c >> 9;               // 0=q 1=k 2=v
    int cc = c & 511;
    int head = cc >> 5, d = cc & 31;
    #pragma unroll
    for (int mt = 0; mt < 2; mt++) {
      #pragma unroll
      for (int r = 0; r < 8; r++) {
        int gm = m0 + mt * 16 + r + hi * 8;
        int w = gm >> 6, m = gm & 63;
        float val = acc[mt][nt][r] + bia;
        size_t wh = (size_t)w * NHD + head;
        if (which == 0)      qh[(wh * NPAD + m) * HD + d] = (_Float16)(val * SCALE);
        else if (which == 1) kh[(wh * NPAD + m) * HD + d] = (_Float16)val;
        else                 vth[(wh * HD + d) * NPAD + m] = (_Float16)val;  // transposed
      }
    }
  }
}

// ---------------- attention: one block per (window, head) ----------------
__global__ __launch_bounds__(128) void attn_kernel(
    const _Float16* __restrict__ qh, const _Float16* __restrict__ kh,
    const _Float16* __restrict__ vth, const float* __restrict__ rpb,
    _Float16* __restrict__ aow) {
  __shared__ float    S[NPAD * NPAD];
  __shared__ _Float16 P[NPAD * NPAD];
  int wh = blockIdx.x;                 // window*16 + head
  int w = wh >> 4, h = wh & 15;
  int wl = w & 63;
  int wv = threadIdx.x >> 5, L = threadIdx.x & 31, lc = L & 15, hi = L >> 4;
  int m0 = wv * 16;
  const _Float16* qbase = qh  + (size_t)wh * NPAD * HD;
  const _Float16* kbase = kh  + (size_t)wh * NPAD * HD;
  const _Float16* vbase = vth + (size_t)wh * HD * NPAD;

  // S = (Q*scale) @ K^T  — K row-major is directly the B operand of Q.K^T
  const _Float16* qrow = qbase + (m0 + lc) * HD + hi * 16;
  v16h af = ld_afrag(qrow);
  v8f s[4];
  #pragma unroll
  for (int nt = 0; nt < 4; nt++) {
    #pragma unroll
    for (int r = 0; r < 8; r++) s[nt][r] = 0.0f;
    v16h bf = ld_b32B(kbase + (nt * 16 + lc) * HD + hi * 16);
    s[nt] = wmma_f16(af, bf, s[nt]);
  }
  // bias + shift-window mask, pads -> -inf
  int wr7 = (wl >> 3) * WSZ, wc7 = (wl & 7) * WSZ;
  #pragma unroll
  for (int nt = 0; nt < 4; nt++) {
    int j = nt * 16 + lc;
    #pragma unroll
    for (int r = 0; r < 8; r++) {
      int i = m0 + r + hi * 8;
      float val;
      if (i < NTOK && j < NTOK) {
        int ri = i / WSZ, ci = i - ri * WSZ;
        int rj = j / WSZ, cj = j - rj * WSZ;
        int rel = (ri - rj + WSZ - 1) * (2 * WSZ - 1) + (ci - cj + WSZ - 1);
        val = s[nt][r] + rpb[rel * NHD + h];
        int gri = wr7 + ri, gci = wc7 + ci, grj = wr7 + rj, gcj = wc7 + cj;
        int cri = (gri < HH - WSZ) ? 0 : ((gri < HH - SSH) ? 1 : 2);
        int cci = (gci < HH - WSZ) ? 0 : ((gci < HH - SSH) ? 1 : 2);
        int crj = (grj < HH - WSZ) ? 0 : ((grj < HH - SSH) ? 1 : 2);
        int ccj = (gcj < HH - WSZ) ? 0 : ((gcj < HH - SSH) ? 1 : 2);
        if (cri * 3 + cci != crj * 3 + ccj) val -= 100.0f;
      } else {
        val = -1e30f;                  // padded rows/cols
      }
      S[i * NPAD + j] = val;
    }
  }
  __syncthreads();
  // softmax per row (64 rows -> threads 0..63), with max(p, 1e-8) clamp
  int t = threadIdx.x;
  if (t < NPAD) {
    float* row = S + t * NPAD;
    float mx = -3.0e38f;
    for (int j = 0; j < NPAD; j++) mx = fmaxf(mx, row[j]);
    float sum = 0.f;
    for (int j = 0; j < NPAD; j++) { float e = __expf(row[j] - mx); row[j] = e; sum += e; }
    float inv = 1.0f / sum;
    _Float16* prow = P + t * NPAD;
    for (int j = 0; j < NPAD; j++) prow[j] = (_Float16)fmaxf(row[j] * inv, 1e-8f);
  }
  __syncthreads();
  // O = P @ V  — V stored transposed, so it is directly the B operand
  #pragma unroll
  for (int dt = 0; dt < 2; dt++) {
    v8f o;
    #pragma unroll
    for (int r = 0; r < 8; r++) o[r] = 0.0f;
    #pragma unroll
    for (int j0 = 0; j0 < NPAD; j0 += 32) {
      v16h ap = ld_afrag(P + (m0 + lc) * NPAD + j0 + hi * 16);
      v16h bf = ld_b32B(vbase + (dt * 16 + lc) * NPAD + j0 + hi * 16);
      o = wmma_f16(ap, bf, o);
    }
    int c = h * HD + dt * 16 + lc;
    #pragma unroll
    for (int r = 0; r < 8; r++) {
      int m = m0 + r + hi * 8;
      aow[((size_t)w * NPAD + m) * CC + c] = (_Float16)o[r];
    }
  }
}

// ---------------- GEMM: proj + window-reverse + unshift + residual (f32 out) ----------------
__global__ __launch_bounds__(128) void gemm_proj(
    const _Float16* __restrict__ X, const _Float16* __restrict__ Wt,
    const float* __restrict__ bias, const float* __restrict__ xin,
    float* __restrict__ x2) {
  int wv = threadIdx.x >> 5, L = threadIdx.x & 31, lc = L & 15, hi = L >> 4;
  int m0 = blockIdx.x * 128 + wv * 32, n0 = blockIdx.y * 64;
  v8f acc[2][4];
  gemm_core(X, Wt, CC, m0, n0, lc, hi, acc);
  #pragma unroll
  for (int nt = 0; nt < 4; nt++) {
    int c = n0 + nt * 16 + lc;
    float bia = bias[c];
    #pragma unroll
    for (int mt = 0; mt < 2; mt++) {
      #pragma unroll
      for (int r = 0; r < 8; r++) {
        int gm = m0 + mt * 16 + r + hi * 8;
        int w = gm >> 6, m = gm & 63;
        if (m < NTOK) {
          int wl = w & 63, b = w >> 6;
          int mr = m / WSZ, mc = m - mr * WSZ;
          int gr = (wl >> 3) * WSZ + mr, gc = (wl & 7) * WSZ + mc;
          int hh = gr + SSH; if (hh >= HH) hh -= HH;
          int ww = gc + SSH; if (ww >= HH) ww -= HH;
          size_t idx = ((size_t)b * (HH*HH) + hh * HH + ww) * CC + c;
          x2[idx] = xin[idx] + acc[mt][nt][r] + bia;
        }
      }
    }
  }
}

// ---------------- GEMM: FC1 + exact GELU (f16 out) ----------------
__global__ __launch_bounds__(128) void gemm_fc1(
    const _Float16* __restrict__ X, const _Float16* __restrict__ Wt,
    const float* __restrict__ bias, _Float16* __restrict__ hid) {
  int wv = threadIdx.x >> 5, L = threadIdx.x & 31, lc = L & 15, hi = L >> 4;
  int m0 = blockIdx.x * 128 + wv * 32, n0 = blockIdx.y * 64;
  v8f acc[2][4];
  gemm_core(X, Wt, CC, m0, n0, lc, hi, acc);
  #pragma unroll
  for (int nt = 0; nt < 4; nt++) {
    int c = n0 + nt * 16 + lc;
    float bia = bias[c];
    #pragma unroll
    for (int mt = 0; mt < 2; mt++) {
      #pragma unroll
      for (int r = 0; r < 8; r++) {
        int gm = m0 + mt * 16 + r + hi * 8;
        float v = acc[mt][nt][r] + bia;
        float gel = 0.5f * v * (1.0f + erff(v * 0.70710678118654752f));
        hid[(size_t)gm * MLPD + c] = (_Float16)gel;
      }
    }
  }
}

// ---------------- GEMM: FC2 + residual -> final f32 output ----------------
__global__ __launch_bounds__(128) void gemm_fc2(
    const _Float16* __restrict__ X, const _Float16* __restrict__ Wt,
    const float* __restrict__ bias, const float* __restrict__ x2,
    float* __restrict__ out) {
  int wv = threadIdx.x >> 5, L = threadIdx.x & 31, lc = L & 15, hi = L >> 4;
  int m0 = blockIdx.x * 128 + wv * 32, n0 = blockIdx.y * 64;
  v8f acc[2][4];
  gemm_core(X, Wt, MLPD, m0, n0, lc, hi, acc);
  #pragma unroll
  for (int nt = 0; nt < 4; nt++) {
    int c = n0 + nt * 16 + lc;
    float bia = bias[c];
    #pragma unroll
    for (int mt = 0; mt < 2; mt++) {
      #pragma unroll
      for (int r = 0; r < 8; r++) {
        int gm = m0 + mt * 16 + r + hi * 8;
        size_t idx = (size_t)gm * CC + c;
        out[idx] = x2[idx] + acc[mt][nt][r] + bia;
      }
    }
  }
}

// ---------------- host orchestration ----------------
extern "C" void kernel_launch(void* const* d_in, const int* in_sizes, int n_in,
                              void* d_out, int out_size, void* d_ws, size_t ws_size,
                              hipStream_t stream) {
  const float* x      = (const float*)d_in[0];
  const float* n1g    = (const float*)d_in[1];
  const float* n1b    = (const float*)d_in[2];
  const float* qkv_w  = (const float*)d_in[3];
  const float* qkv_b  = (const float*)d_in[4];
  const float* proj_w = (const float*)d_in[5];
  const float* proj_b = (const float*)d_in[6];
  const float* rpb    = (const float*)d_in[7];
  const float* n2g    = (const float*)d_in[8];
  const float* n2b    = (const float*)d_in[9];
  const float* w1     = (const float*)d_in[10];
  const float* b1     = (const float*)d_in[11];
  const float* w2     = (const float*)d_in[12];
  const float* b2     = (const float*)d_in[13];
  float* out = (float*)d_out;

  size_t off = 0;
  auto take = [&](size_t bytes) -> void* {
    void* r = (char*)d_ws + off;
    off += (bytes + 255) & ~(size_t)255;
    return r;
  };
  _Float16* wqkvT = (_Float16*)take((size_t)(3*CC) * CC * 2);   // 1536 x 512
  _Float16* wprojT= (_Float16*)take((size_t)CC * CC * 2);       // 512 x 512
  _Float16* w1T   = (_Float16*)take((size_t)MLPD * CC * 2);     // 2048 x 512
  _Float16* w2T   = (_Float16*)take((size_t)CC * MLPD * 2);     // 512 x 2048
  _Float16* xw    = (_Float16*)take((size_t)MPAD * CC * 2);     // 65536 x 512
  _Float16* qh    = (_Float16*)take((size_t)NW_TOT * NHD * NPAD * HD * 2);
  _Float16* kh    = (_Float16*)take((size_t)NW_TOT * NHD * NPAD * HD * 2);
  _Float16* vth   = (_Float16*)take((size_t)NW_TOT * NHD * HD * NPAD * 2);
  float*    x2    = (float*)   take((size_t)MDEN * CC * 4);
  _Float16* hid   = (_Float16*)take((size_t)MDEN * MLPD * 2);
  _Float16* aow   = xw;   // alias: xw is dead once QKV GEMM has consumed it
  _Float16* h2    = qh;   // alias: qh is dead once attention has consumed it

  // weights -> f16 transposed (L2-resident thereafter)
  transpose_cast<<<(CC*3*CC + 255)/256, 256, 0, stream>>>(qkv_w, wqkvT, CC, 3*CC);
  transpose_cast<<<(CC*CC   + 255)/256, 256, 0, stream>>>(proj_w, wprojT, CC, CC);
  transpose_cast<<<(CC*MLPD + 255)/256, 256, 0, stream>>>(w1, w1T, CC, MLPD);
  transpose_cast<<<(MLPD*CC + 255)/256, 256, 0, stream>>>(w2, w2T, MLPD, CC);

  // LN1 + roll(-3,-3) + window partition + pad
  ln_shift_win<<<MPAD/8, 256, 0, stream>>>(x, n1g, n1b, xw);

  // QKV projection
  gemm_qkv<<<dim3(MPAD/128, (3*CC)/64), 128, 0, stream>>>(xw, wqkvT, qkv_b, qh, kh, vth);

  // attention per (window, head)
  attn_kernel<<<NW_TOT * NHD, 128, 0, stream>>>(qh, kh, vth, rpb, aow);

  // output projection + window reverse + roll back + residual
  gemm_proj<<<dim3(MPAD/128, CC/64), 128, 0, stream>>>(aow, wprojT, proj_b, x, x2);

  // LN2
  ln_dense<<<MDEN/8, 256, 0, stream>>>(x2, n2g, n2b, h2);

  // MLP
  gemm_fc1<<<dim3(MDEN/128, MLPD/64), 128, 0, stream>>>(h2, w1T, b1, hid);
  gemm_fc2<<<dim3(MDEN/128, CC/64),  128, 0, stream>>>(hid, w2T, b2, x2, out);
}
